// Recurrent_19636590477757
// MI455X (gfx1250) — matrix operand order repaired and checked by reading, not compiled, not run on hardware
//
#include <hip/hip_runtime.h>
#include <hip/hip_bf16.h>

// Problem constants (reference: B=64, S=2048, D=128, H=128)
#define BB 64
#define SS 2048
#define DD 128
#define HH 128

typedef __bf16 v16bf __attribute__((ext_vector_type(16)));
typedef __bf16 v8bf  __attribute__((ext_vector_type(8)));
typedef __bf16 v2bf  __attribute__((ext_vector_type(2)));
typedef float  v8f   __attribute__((ext_vector_type(8)));
typedef float  f4    __attribute__((ext_vector_type(4)));

// float -> bf16: let the backend use the native convert (fptrunc lowering
// reaches v_cvt_pk_bf16_f32-class ops on gfx1250) instead of manual bit math.
__device__ __forceinline__ __bf16 f2bf(float f) {
    return (__bf16)f;
}

// Pack two f32 -> bf16x2 (single packed convert when available)
__device__ __forceinline__ v2bf pk2bf(float a, float b) {
#if __has_builtin(__builtin_amdgcn_cvt_pk_bf16_f32)
    return __builtin_amdgcn_cvt_pk_bf16_f32(a, b);
#else
    v2bf r; r[0] = (__bf16)a; r[1] = (__bf16)b; return r;
#endif
}

// Branchless tanh: hardware v_tanh_f32 (confirmed present on gfx1250).
__device__ __forceinline__ float fast_tanh(float z) {
#if __has_builtin(__builtin_amdgcn_tanhf)
    return __builtin_amdgcn_tanhf(z);
#else
    float t = __builtin_amdgcn_exp2f(z * 2.885390081777927f); // 2*log2(e)
    return 1.0f - 2.0f * __builtin_amdgcn_rcpf(t + 1.0f);
#endif
}

// Split workgroup barrier that only drains LDS (DScnt). Unlike __syncthreads()
// it does NOT wait for outstanding global stores (storecnt), keeping the 8
// output stores per step in flight across iterations. Safe per ISA: a store's source
// GPRs may be overwritten once it is issued.
__device__ __forceinline__ void lds_barrier() {
    asm volatile("s_wait_dscnt 0x0\n\t"
                 "s_barrier_signal -1\n\t"
                 "s_barrier_wait -1" ::: "memory");
}

// Build a 16x32 bf16 A-fragment from a row-major [16][128] bf16 LDS tile.
// Per CDNA5 ISA 7.12.2: lane L holds row M=L%16; lanes 0-15 carry K in
// [kc*32+0..7] and [kc*32+16..23]; lanes 16-31 carry K in [+8..15],[+24..31].
__device__ __forceinline__ v16bf load_a_frag(const __bf16* lds, int lm, int lh, int kc) {
    const int k1 = kc * 32 + lh * 8;
    v8bf lo = *(const v8bf*)(lds + lm * 128 + k1);
    v8bf hi = *(const v8bf*)(lds + lm * 128 + k1 + 16);
    return __builtin_shufflevector(lo, hi, 0,1,2,3,4,5,6,7,8,9,10,11,12,13,14,15);
}

// Cooperative stage of one 16x128 f32 x-tile into LDS as bf16 (256 threads).
__device__ __forceinline__ void stage_x(const float* __restrict__ x,
                                        __bf16* __restrict__ dst,
                                        int tid, int wgb, int s) {
    const int flat = tid * 8;            // 256 threads * 8 = 2048 elems
    const int m = flat >> 7;
    const int d = flat & 127;
    const float* src = x + ((size_t)(wgb + m) * SS + s) * DD + d;
    f4 lo = *(const f4*)(src);
    f4 hi = *(const f4*)(src + 4);
    v8bf v;
#pragma unroll
    for (int i = 0; i < 2; ++i) {
        v2bf p0 = pk2bf(lo[2 * i], lo[2 * i + 1]);
        v2bf p1 = pk2bf(hi[2 * i], hi[2 * i + 1]);
        v[2 * i]     = p0[0];
        v[2 * i + 1] = p0[1];
        v[2 * i + 4] = p1[0];
        v[2 * i + 5] = p1[1];
    }
    *(v8bf*)(dst + m * 128 + d) = v;
}

__global__ __launch_bounds__(256)
void Recurrent_19636590477757_kernel(const float* __restrict__ x,
                                     const float* __restrict__ W,
                                     const float* __restrict__ bias,
                                     float* __restrict__ out) {
    // out layout: outputs (B,S,H) flat, then hidden (B,H) flat
    float* __restrict__ hidden = out + (size_t)BB * SS * HH;

    // Ping-pong buffers: one barrier per step.
    __shared__ __bf16 lds_h[2][16 * 128];
    __shared__ __bf16 lds_x[2][16 * 128];

    const int tid  = threadIdx.x;
    const int wave = tid >> 5;        // 0..7 -> output-column tile
    const int lane = tid & 31;
    const int lm   = lane & 15;       // lane-in-half
    const int lh   = lane >> 4;       // half select
    const int wgb  = blockIdx.x * 16; // this workgroup's 16 batch rows
    const int ncol = wave * 16 + lm;  // B/C/D column owned by this lane

    // ---- Preload B-fragments of Wx and Wh into registers (per ISA 32x16 bf16
    // B layout: lane L -> column N=L%16, K = kc*32 + 8*(L/16) + i (i<8), +16 for i>=8). ----
    v16bf wxf[4], whf[4];
#pragma unroll
    for (int kc = 0; kc < 4; ++kc) {
        v16bf bx, bh;
#pragma unroll
        for (int i = 0; i < 16; ++i) {
            int k = kc * 32 + lh * 8 + (i < 8 ? i : i + 8);
            bx[i] = f2bf(W[(size_t)k * HH + ncol]);          // Wx = W[0:D]
            bh[i] = f2bf(W[(size_t)(DD + k) * HH + ncol]);   // Wh = W[D:D+H]
        }
        wxf[kc] = bx;
        whf[kc] = bh;
    }
    const float bv = bias[ncol];

    // ---- Init: zero h[0], stage x_0 into buffer 0 ----
    {
        const int flat = tid * 8;
        const int m = flat >> 7;
        const int d = flat & 127;
        v8bf z = {};
        *(v8bf*)(&lds_h[0][m * 128 + d]) = z;
        stage_x(x, lds_x[0], tid, wgb, 0);
    }
    __syncthreads();

    // ---- Sequential scan: one LDS-only barrier per step ----
    for (int s = 0; s < SS; ++s) {
        const int cur = s & 1;
        const int nxt = cur ^ 1;

        // Hoist all 16 ds_load_b128 so DScnt waits pipeline instead of
        // draining to zero before every WMMA.
        v16bf ax[4], ah[4];
#pragma unroll
        for (int kc = 0; kc < 4; ++kc) {
            ax[kc] = load_a_frag(lds_x[cur], lm, lh, kc);
            ah[kc] = load_a_frag(lds_h[cur], lm, lh, kc);
        }

        // Two independent accumulator chains (depth 4 each, interleavable).
        v8f accx; // x_t @ Wx + b
#pragma unroll
        for (int r = 0; r < 8; ++r) accx[r] = bv;
        v8f acch = {}; // h_{t-1} @ Wh
#pragma unroll
        for (int kc = 0; kc < 4; ++kc) {
            accx = __builtin_amdgcn_wmma_f32_16x16x32_bf16(
                false, ax[kc], false, wxf[kc], (short)0, accx, false, false);
            acch = __builtin_amdgcn_wmma_f32_16x16x32_bf16(
                false, ah[kc], false, whf[kc], (short)0, acch, false, false);
        }

        // Prefetch x_{s+1} into the other buffer (no reader conflict).
        if (s + 1 < SS) stage_x(x, lds_x[nxt], tid, wgb, s + 1);

        // h_new = tanh(accx + acch); C/D layout: VGPR r -> row M = r + 8*lh.
        float hv[8];
#pragma unroll
        for (int r = 0; r < 8; ++r) hv[r] = fast_tanh(accx[r] + acch[r]);

#pragma unroll
        for (int r = 0; r < 8; r += 2) {
            v2bf p = pk2bf(hv[r], hv[r + 1]);
            lds_h[nxt][(r + 8 * lh) * 128 + ncol]     = p[0];
            lds_h[nxt][(r + 1 + 8 * lh) * 128 + ncol] = p[1];
        }
#pragma unroll
        for (int r = 0; r < 8; ++r) {
            const int m = r + 8 * lh;
            __builtin_nontemporal_store(hv[r], &out[((size_t)(wgb + m) * SS + s) * HH + ncol]);
        }
        if (s == SS - 1) {
#pragma unroll
            for (int r = 0; r < 8; ++r) {
                const int m = r + 8 * lh;
                hidden[(size_t)(wgb + m) * HH + ncol] = hv[r];
            }
        }
        // Publish h_new / x_{s+1}; orders this step's LDS reads before next
        // step's overwrites. Global stores intentionally left in flight.
        lds_barrier();
    }
}

extern "C" void kernel_launch(void* const* d_in, const int* in_sizes, int n_in,
                              void* d_out, int out_size, void* d_ws, size_t ws_size,
                              hipStream_t stream) {
    const float* x = (const float*)d_in[0]; // (B,S,D)
    const float* W = (const float*)d_in[1]; // (D+H,H)
    const float* b = (const float*)d_in[2]; // (H,)
    float* out = (float*)d_out;             // (B,S,H) ++ (B,H)
    (void)in_sizes; (void)n_in; (void)out_size; (void)d_ws; (void)ws_size;

    Recurrent_19636590477757_kernel<<<dim3(BB / 16), dim3(256), 0, stream>>>(x, W, b, out);
}
